// multi_head_attention_7155415515667
// MI455X (gfx1250) — matrix operand order repaired and checked by reading, not compiled, not run on hardware
//
#include <hip/hip_runtime.h>
#include <hip/hip_bf16.h>

typedef _Float16 h16;
typedef __attribute__((ext_vector_type(2)))  __fp16   v2fp;   // return type of cvt_pkrtz
typedef __attribute__((ext_vector_type(8)))  _Float16 v8h;
typedef __attribute__((ext_vector_type(16))) _Float16 v16h;
typedef __attribute__((ext_vector_type(8)))  float    v8f;

#define N_SEQ  2048
#define BATCH  2
#define HEADS_ 8
#define EDIM   64
#define FDIM   256   // feat / query dim
#define CATDIM 512   // HEADS_*EDIM
#define ODIM   256

union V16U { v16h v; v8h h[2]; unsigned int u[8]; };

static __device__ __forceinline__ v8f wmma_f16(v16h a, v16h b, v8f c) {
  // D = A(16x32 f16) * B(32x16 f16) + C(16x16 f32)
  return __builtin_amdgcn_wmma_f32_16x16x32_f16(false, a, false, b, (short)0, c, false, false);
}

// A-matrix 16x32 fragment: lane m = l%16 is the M row; rowbase already points
// at row m, element k0. elems i = K half*16+i.
static __device__ __forceinline__ v16h load_a16x32(const h16* rowbase, int half) {
  return *(const v16h*)(rowbase + half * 16);
}

// B-matrix 32x16 fragment: lane n = l%16 is the N column; rowbase points at the
// contiguous K-run for this lane's column. elems i = K half*16+i.
static __device__ __forceinline__ v16h load_b32x16(const h16* rowbase, int half) {
  return *(const v16h*)(rowbase + half * 16);
}

static __device__ __forceinline__ unsigned int pack2_f16(float a, float b) {
  union { v2fp h; unsigned int u; } r;
  r.h = __builtin_amdgcn_cvt_pkrtz(a, b);
  return r.u;
}

// ---------------------------------------------------------------- casts
__global__ void cast_f16_kernel(const float* __restrict__ src, h16* __restrict__ dst, int n) {
  int i = blockIdx.x * blockDim.x + threadIdx.x;
  if (i < n) dst[i] = (h16)src[i];
}

// src[bn][R][C] (f32) -> dst[bn][C][R] (f16)
__global__ void transpose_cast_kernel(const float* __restrict__ src, h16* __restrict__ dst,
                                      int R, int C, int total) {
  int i = blockIdx.x * blockDim.x + threadIdx.x;
  if (i >= total) return;
  int bn = i / (R * C);
  int rem = i - bn * R * C;
  int c = rem / R;
  int r = rem - c * R;
  dst[i] = (h16)src[(size_t)bn * R * C + (size_t)r * C + c];
}

// ---------------------------------------------------------------- Q/K projection
// Y[b,h][n][e] = (X[b][n][:] @ W[h] + bias[h]) * scale    (f16 out, row-major)
// Computed as D = Wt(A: 16e x 32d) * X^T(B: lane=n)  -> contiguous row stores.
__global__ __launch_bounds__(256) void proj_qk_kernel(
    const h16* __restrict__ X, const h16* __restrict__ Wt,
    const float* __restrict__ bias, h16* __restrict__ Y, float scale) {
  const int lane = threadIdx.x & 31;
  const int wave = threadIdx.x >> 5;
  const int half = lane >> 4;
  const int ln16 = lane & 15;
  const int h = blockIdx.y, b = blockIdx.z;
  const int n0 = blockIdx.x * 128 + wave * 16;

  const h16* xrow = X + ((size_t)b * N_SEQ + n0 + ln16) * FDIM;
  v16h bx[8];
#pragma unroll
  for (int ks = 0; ks < 8; ++ks) bx[ks] = load_b32x16(xrow + ks * 32, half);

  v8f c[4] = {};
  const h16* Wh = Wt + (size_t)h * EDIM * FDIM;
#pragma unroll
  for (int t = 0; t < 4; ++t) {
    const h16* arow = Wh + (size_t)(t * 16 + ln16) * FDIM;
#pragma unroll
    for (int ks = 0; ks < 8; ++ks)
      c[t] = wmma_f16(load_a16x32(arow + ks * 32, half), bx[ks], c[t]);
  }

  const float* bh = bias + h * EDIM;
  h16* yrow = Y + (((size_t)(b * HEADS_ + h)) * N_SEQ + n0 + ln16) * EDIM;
#pragma unroll
  for (int t = 0; t < 4; ++t) {
    v8h r;
#pragma unroll
    for (int j = 0; j < 8; ++j)
      r[j] = (h16)((c[t][j] + bh[t * 16 + half * 8 + j]) * scale);
    *(v8h*)(yrow + t * 16 + half * 8) = r;
  }
}

// ---------------------------------------------------------------- V projection
// Vt[b,h][e][n] = feat[b][n][:] @ Wv[h] + bv[h]   (f16 out, TRANSPOSED)
// Computed as D = X(A: 16n x 32d) * Wt(B: lane=e) -> contiguous transposed stores.
__global__ __launch_bounds__(256) void proj_v_kernel(
    const h16* __restrict__ X, const h16* __restrict__ Wt,
    const float* __restrict__ bias, h16* __restrict__ Yt) {
  const int lane = threadIdx.x & 31;
  const int wave = threadIdx.x >> 5;
  const int half = lane >> 4;
  const int ln16 = lane & 15;
  const int h = blockIdx.y, b = blockIdx.z;
  const int n0 = blockIdx.x * 128 + wave * 16;

  const h16* xrow = X + ((size_t)b * N_SEQ + n0 + ln16) * FDIM;
  v16h ax[8];
#pragma unroll
  for (int ks = 0; ks < 8; ++ks) ax[ks] = load_a16x32(xrow + ks * 32, half);

  v8f c[4] = {};
  const h16* Wh = Wt + (size_t)h * EDIM * FDIM;
#pragma unroll
  for (int t = 0; t < 4; ++t) {
    const h16* brow = Wh + (size_t)(t * 16 + ln16) * FDIM;
#pragma unroll
    for (int ks = 0; ks < 8; ++ks)
      c[t] = wmma_f16(ax[ks], load_b32x16(brow + ks * 32, half), c[t]);
  }

#pragma unroll
  for (int t = 0; t < 4; ++t) {
    const int e = t * 16 + ln16;
    const float bb = bias[h * EDIM + e];
    v8h r;
#pragma unroll
    for (int j = 0; j < 8; ++j) r[j] = (h16)(c[t][j] + bb);
    *(v8h*)(Yt + (((size_t)(b * HEADS_ + h)) * EDIM + e) * N_SEQ + n0 + half * 8) = r;
  }
}

// ---------------------------------------------------------------- flash attention
// Q pre-scaled by 1/sqrt(64). Per wave: 16 queries, online softmax over kv tiles
// of 32. S^T = K*Q^T (so softmax rows live in-lane), heads^T = V^T * P^T.
__global__ __launch_bounds__(256) void flash_attn_kernel(
    const h16* __restrict__ Q, const h16* __restrict__ K,
    const h16* __restrict__ Vt, h16* __restrict__ cat) {
  const int lane = threadIdx.x & 31;
  const int wave = threadIdx.x >> 5;
  const int half = lane >> 4;
  const int ln16 = lane & 15;
  const int h = blockIdx.y, b = blockIdx.z;
  const int q0 = blockIdx.x * 128 + wave * 16;
  const size_t bh = (size_t)(b * HEADS_ + h);

  const h16* Qbh = Q + bh * N_SEQ * EDIM;
  const h16* Kbh = K + bh * N_SEQ * EDIM;
  const h16* Vbh = Vt + bh * (size_t)EDIM * N_SEQ;

  // Q^T B-fragments (K-dim = e, two k-steps of 32)
  v16h qf[2];
  {
    const h16* qrow = Qbh + (size_t)(q0 + ln16) * EDIM;
    qf[0] = load_b32x16(qrow, half);
    qf[1] = load_b32x16(qrow + 32, half);
  }

  v8f o[4] = {};
  float m_run = -__builtin_inff();
  float l_run = 0.0f;

  for (int kv0 = 0; kv0 < N_SEQ; kv0 += 32) {
    // S^T tiles: s0 = kv rows [kv0,kv0+16), s1 = [kv0+16,kv0+32); cols = 16 q's
    v8f s0 = {}, s1 = {};
#pragma unroll
    for (int ks = 0; ks < 2; ++ks) {
      v16h a0 = load_a16x32(Kbh + (size_t)(kv0 + ln16) * EDIM + ks * 32, half);
      v16h a1 = load_a16x32(Kbh + (size_t)(kv0 + 16 + ln16) * EDIM + ks * 32, half);
      s0 = wmma_f16(a0, qf[ks], s0);
      s1 = wmma_f16(a1, qf[ks], s1);
    }

    // online softmax: per lane, 16 kv values for query column ln16; the other
    // 16 kv of this tile live in the partner lane (xor 16).
    float tmax = s0[0];
#pragma unroll
    for (int j = 0; j < 8; ++j) tmax = fmaxf(tmax, fmaxf(s0[j], s1[j]));
    tmax = fmaxf(tmax, __shfl_xor(tmax, 16, 32));
    const float mnew = fmaxf(m_run, tmax);
    const float alpha = __expf(m_run - mnew);
    float p0[8], p1[8], tsum = 0.0f;
#pragma unroll
    for (int j = 0; j < 8; ++j) {
      p0[j] = __expf(s0[j] - mnew);
      p1[j] = __expf(s1[j] - mnew);
      tsum += p0[j] + p1[j];
    }
    tsum += __shfl_xor(tsum, 16, 32);
    l_run = l_run * alpha + tsum;
    m_run = mnew;
#pragma unroll
    for (int t = 0; t < 4; ++t) o[t] = o[t] * alpha;

    // Build P^T B-fragment (32kv x 16q) from the two D-tiles: one cross-half
    // shuffle per packed register.
    unsigned int pk0[4], pk1[4];
#pragma unroll
    for (int i = 0; i < 4; ++i) {
      pk0[i] = pack2_f16(p0[2 * i], p0[2 * i + 1]);
      pk1[i] = pack2_f16(p1[2 * i], p1[2 * i + 1]);
    }
    V16U bp;
#pragma unroll
    for (int i = 0; i < 4; ++i) {
      const unsigned int ox0 = __shfl_xor(pk0[i], 16, 32);
      const unsigned int ox1 = __shfl_xor(pk1[i], 16, 32);
      bp.u[i]     = half ? ox1 : pk0[i];
      bp.u[4 + i] = half ? pk1[i] : ox0;
    }

    // heads^T accumulate: o[t] (16e x 16q) += Vt tile (16e x 32kv) * P^T
#pragma unroll
    for (int t = 0; t < 4; ++t) {
      v16h av = load_a16x32(Vbh + (size_t)(t * 16 + ln16) * N_SEQ + kv0, half);
      o[t] = wmma_f16(av, bp.v, o[t]);
    }
  }

  const float inv = 1.0f / l_run;
  h16* dst = cat + ((size_t)(b * N_SEQ) + q0 + ln16) * CATDIM + h * EDIM;
#pragma unroll
  for (int t = 0; t < 4; ++t) {
    v8h r;
#pragma unroll
    for (int j = 0; j < 8; ++j) r[j] = (h16)(o[t][j] * inv);
    *(v8h*)(dst + t * 16 + half * 8) = r;
  }
}

// ---------------------------------------------------------------- output projection
// out[b][n][o] = cat[b][n][:512] @ Wo + bo   (f32 out)
// D = Wot(A: 16o x 32c) * cat^T(B: lane=n)
__global__ __launch_bounds__(256) void out_proj_kernel(
    const h16* __restrict__ cat, const h16* __restrict__ Wot,
    const float* __restrict__ bo, float* __restrict__ out) {
  const int lane = threadIdx.x & 31;
  const int wave = threadIdx.x >> 5;
  const int half = lane >> 4;
  const int ln16 = lane & 15;
  const int ob = blockIdx.y * 64;
  const int b = blockIdx.z;
  const int n0 = blockIdx.x * 128 + wave * 16;

  const h16* crow = cat + ((size_t)(b * N_SEQ) + n0 + ln16) * CATDIM;
  v8f c[4] = {};
#pragma unroll
  for (int ks = 0; ks < 16; ++ks) {
    v16h bc = load_b32x16(crow + ks * 32, half);
#pragma unroll
    for (int t = 0; t < 4; ++t) {
      v16h aw = load_a16x32(Wot + (size_t)(ob + t * 16 + ln16) * CATDIM + ks * 32, half);
      c[t] = wmma_f16(aw, bc, c[t]);
    }
  }

#pragma unroll
  for (int t = 0; t < 4; ++t) {
    float* drow = out + ((size_t)(b * N_SEQ) + n0 + ln16) * ODIM + ob + t * 16 + half * 8;
    v8f r;
#pragma unroll
    for (int j = 0; j < 8; ++j) r[j] = c[t][j] + bo[ob + t * 16 + half * 8 + j];
    *(v8f*)drow = r;
  }
}

// ---------------------------------------------------------------- launch
extern "C" void kernel_launch(void* const* d_in, const int* in_sizes, int n_in,
                              void* d_out, int out_size, void* d_ws, size_t ws_size,
                              hipStream_t stream) {
  (void)in_sizes; (void)n_in; (void)out_size; (void)ws_size;
  const float* feat  = (const float*)d_in[0];
  const float* query = (const float*)d_in[1];
  const float* Wq = (const float*)d_in[2];
  const float* bq = (const float*)d_in[3];
  const float* Wk = (const float*)d_in[4];
  const float* bk = (const float*)d_in[5];
  const float* Wv = (const float*)d_in[6];
  const float* bv = (const float*)d_in[7];
  const float* Wo = (const float*)d_in[8];
  const float* bo = (const float*)d_in[9];
  float* out = (float*)d_out;

  // workspace carve-up (all f16, sizes are powers of two -> aligned)
  h16* p = (h16*)d_ws;
  h16* query16 = p; p += (size_t)BATCH * N_SEQ * FDIM;          // 1M elems
  h16* feat16  = p; p += (size_t)BATCH * N_SEQ * FDIM;
  h16* Wq_t = p; p += (size_t)HEADS_ * EDIM * FDIM;             // 128K
  h16* Wk_t = p; p += (size_t)HEADS_ * EDIM * FDIM;
  h16* Wv_t = p; p += (size_t)HEADS_ * EDIM * FDIM;
  h16* Wo_t = p; p += (size_t)ODIM * CATDIM;                    // 128K
  h16* Q16  = p; p += (size_t)BATCH * HEADS_ * N_SEQ * EDIM;    // 2M
  h16* K16  = p; p += (size_t)BATCH * HEADS_ * N_SEQ * EDIM;
  h16* Vt16 = p; p += (size_t)BATCH * HEADS_ * EDIM * N_SEQ;
  h16* cat16 = p;                                               // 2M

  const int nIn = BATCH * N_SEQ * FDIM;
  cast_f16_kernel<<<(nIn + 255) / 256, 256, 0, stream>>>(query, query16, nIn);
  cast_f16_kernel<<<(nIn + 255) / 256, 256, 0, stream>>>(feat, feat16, nIn);

  const int wTot = HEADS_ * FDIM * EDIM;  // 131072
  transpose_cast_kernel<<<(wTot + 255) / 256, 256, 0, stream>>>(Wq, Wq_t, FDIM, EDIM, wTot);
  transpose_cast_kernel<<<(wTot + 255) / 256, 256, 0, stream>>>(Wk, Wk_t, FDIM, EDIM, wTot);
  transpose_cast_kernel<<<(wTot + 255) / 256, 256, 0, stream>>>(Wv, Wv_t, FDIM, EDIM, wTot);
  const int woTot = CATDIM * ODIM;        // 131072
  transpose_cast_kernel<<<(woTot + 255) / 256, 256, 0, stream>>>(Wo, Wo_t, CATDIM, ODIM, woTot);

  dim3 gProj(N_SEQ / 128, HEADS_, BATCH);
  proj_qk_kernel<<<gProj, 256, 0, stream>>>(query16, Wq_t, bq, Q16, 0.125f);  // fold 1/sqrt(64)
  proj_qk_kernel<<<gProj, 256, 0, stream>>>(feat16, Wk_t, bk, K16, 1.0f);
  proj_v_kernel<<<gProj, 256, 0, stream>>>(feat16, Wv_t, bv, Vt16);

  flash_attn_kernel<<<gProj, 256, 0, stream>>>(Q16, K16, Vt16, cat16);

  dim3 gOut(N_SEQ / 128, ODIM / 64, BATCH);
  out_proj_kernel<<<gOut, 256, 0, stream>>>(cat16, Wo_t, bo, out);
}
